// GVANet_45217415693011
// MI455X (gfx1250) — matrix operand order repaired and checked by reading, not compiled, not run on hardware
//
#include <hip/hip_runtime.h>

typedef __attribute__((ext_vector_type(16))) _Float16 v16h;
typedef __attribute__((ext_vector_type(8)))  float    v8f;

#define EPS 1e-5f

struct BnP { const float *g, *b, *m, *v; };

struct EdgeArgs {
  const float* xin;  long xin_bstride;     // [B][C][2048], batch stride in floats
  const int*   idx;                        // [B][2048][32]
  BnP bnin;
  const float* w[4];
  BnP bn[4];
  float* out; long out_bstride; int out_coff;  // cat buffer [B][128][2048]
  _Float16* out16;                         // optional [B][64][2048]
};

struct ClsArgs {
  const float* cat;                        // [B][128][2048]
  const float* w[3];
  const float* bias[3];
  BnP bn[3];
  float* y;                                // [B*40][2048] per-sample activations
};

// ---------------------------------------------------------------------------
// kNN over C=3 input (exact f32). One thread per point, batch points in LDS.
// Register-resident sorted top-32 (ascending), ties keep earliest index.
// ---------------------------------------------------------------------------
__global__ void __launch_bounds__(256) knn3_kernel(const float* __restrict__ x0,
                                                   int* __restrict__ idxout) {
  __shared__ float sx[3 * 2048];
  int b = blockIdx.y;
  for (int e = threadIdx.x; e < 3 * 2048; e += 256)
    sx[e] = x0[(long)b * 3 * 2048 + e];
  __syncthreads();
  int n = blockIdx.x * 256 + threadIdx.x;
  float a0 = sx[n], a1 = sx[2048 + n], a2 = sx[4096 + n];
  float bd[32]; int bi[32];
#pragma unroll
  for (int j = 0; j < 32; ++j) { bd[j] = 3.4e38f; bi[j] = 0; }
  for (int m = 0; m < 2048; ++m) {
    float d0 = sx[m] - a0, d1 = sx[2048 + m] - a1, d2 = sx[4096 + m] - a2;
    float d = d0 * d0 + d1 * d1 + d2 * d2;
    if (d < bd[31]) {
      bd[31] = d; bi[31] = m;
#pragma unroll
      for (int j = 31; j > 0; --j)
        if (bd[j] < bd[j - 1]) {
          float td = bd[j]; bd[j] = bd[j - 1]; bd[j - 1] = td;
          int   ti = bi[j]; bi[j] = bi[j - 1]; bi[j - 1] = ti;
        }
    }
  }
  int* o = idxout + ((long)b * 2048 + n) * 32;
#pragma unroll
  for (int j = 0; j < 32; ++j) o[j] = bi[j];
}

// ---------------------------------------------------------------------------
// Gram matrix S = X^T X for one batch, X = [64][2048] f16. One wave per 16x16
// output tile, K = 64 channels = 2 WMMA steps.
// ---------------------------------------------------------------------------
__global__ void __launch_bounds__(256) gram_kernel(const _Float16* __restrict__ X,
                                                   float* __restrict__ S) {
  int wave = threadIdx.x >> 5, lane = threadIdx.x & 31;
  int tile = blockIdx.x * 8 + wave;             // 128*128 tiles, grid = 2048
  int it = tile >> 7, jt = tile & 127;
  int half = (lane >> 4) & 1, lm = lane & 15;
  int i0 = it << 4, j0 = jt << 4;
  v8f acc = {};
  for (int kb = 0; kb < 2; ++kb) {
    v16h a, bf;
#pragma unroll
    for (int h = 0; h < 16; ++h) {
      int kk = (kb << 5) + half * 16 + h;
      a[h] = X[kk * 2048 + i0 + lm];
    }
#pragma unroll
    for (int h = 0; h < 16; ++h) {
      int kk = (kb << 5) + half * 16 + h;
      bf[h] = X[kk * 2048 + j0 + lm];
    }
    acc = __builtin_amdgcn_wmma_f32_16x16x32_f16(false, a, false, bf,
                                                 (short)0, acc, false, false);
  }
#pragma unroll
  for (int r = 0; r < 8; ++r) {
    int mi = half * 8 + r;
    S[(long)(i0 + mi) * 2048 + j0 + lm] = acc[r];
  }
}

// ---------------------------------------------------------------------------
// Top-32 selection from a Gram row: d = xx[n] + xx[m] - 2*S[n][m].
// ---------------------------------------------------------------------------
__global__ void __launch_bounds__(256) knn_from_gram(const float* __restrict__ S,
                                                     const float* __restrict__ xx,
                                                     int b, int* __restrict__ idxout) {
  int n = blockIdx.x * 256 + threadIdx.x;
  const float* row = S + (long)n * 2048;
  const float* xxb = xx + b * 2048;
  float xn = xxb[n];
  float bd[32]; int bi[32];
#pragma unroll
  for (int j = 0; j < 32; ++j) { bd[j] = 3.4e38f; bi[j] = 0; }
  for (int m = 0; m < 2048; ++m) {
    float d = xn + xxb[m] - 2.0f * row[m];
    if (d < bd[31]) {
      bd[31] = d; bi[31] = m;
#pragma unroll
      for (int j = 31; j > 0; --j)
        if (bd[j] < bd[j - 1]) {
          float td = bd[j]; bd[j] = bd[j - 1]; bd[j - 1] = td;
          int   ti = bi[j]; bi[j] = bi[j - 1]; bi[j - 1] = ti;
        }
    }
  }
  int* o = idxout + ((long)b * 2048 + n) * 32;
#pragma unroll
  for (int j = 0; j < 32; ++j) o[j] = bi[j];
}

// Sum of squares of x1 (= cat channels 0..63).
__global__ void __launch_bounds__(256) sumsq_kernel(const float* __restrict__ cat,
                                                    float* __restrict__ xx) {
  int i = blockIdx.x * 256 + threadIdx.x;       // 16*2048
  int b = i >> 11, n = i & 2047;
  const float* p = cat + (long)b * 128 * 2048 + n;
  float s = 0.f;
  for (int c = 0; c < 64; ++c) { float v = p[c * 2048]; s += v * v; }
  xx[i] = s;
}

// ---------------------------------------------------------------------------
// Generic conv-along-k as WMMA GEMM. Activations in LDS f16, channel stride
// 512, samples packed n*len+j for 16 points. K-dim = CINL*KW, kk = i*KW+t
// (matches the (O,I,1,KW) weight layout exactly, so A-frag loads are
// contiguous). BN+ReLU fused on the f32 accumulator at store.
// ---------------------------------------------------------------------------
__device__ __forceinline__ void conv_wmma(const _Float16* __restrict__ Hin,
                                          _Float16* __restrict__ Hout,
                                          const float* __restrict__ W,
                                          int CINL, int KW, int LGKW,
                                          int len_in, int len_out, BnP bn) {
  int tid = threadIdx.x;
  int wave = tid >> 5, lane = tid & 31;
  int half = (lane >> 4) & 1, lm = lane & 15;
  const int NT = len_out;                        // (16*len_out)/16
  const int KTOT = CINL * KW;
  const int KB = (KTOT + 31) >> 5;
  for (int t = wave; t < 4 * NT; t += 8) {
    int mt = t / NT, nt = t - mt * NT;
    int col = (nt << 4) | lm;
    int n_pt = col / len_out, jj = col - n_pt * len_out;
    v8f acc = {};
    for (int kb = 0; kb < KB; ++kb) {
      v16h a, bf;
      int o = (mt << 4) | lm;
#pragma unroll
      for (int h = 0; h < 16; ++h) {
        int kk = (kb << 5) + half * 16 + h;
        a[h] = (kk < KTOT) ? (_Float16)W[o * KTOT + kk] : (_Float16)0.0f;
      }
#pragma unroll
      for (int h = 0; h < 16; ++h) {
        int kk = (kb << 5) + half * 16 + h;
        _Float16 vh = (_Float16)0.0f;
        if (kk < KTOT) {
          int ic = kk >> LGKW;
          int tt = kk & (KW - 1);
          vh = Hin[ic * 512 + n_pt * len_in + jj + tt];
        }
        bf[h] = vh;
      }
      acc = __builtin_amdgcn_wmma_f32_16x16x32_f16(false, a, false, bf,
                                                   (short)0, acc, false, false);
    }
#pragma unroll
    for (int r = 0; r < 8; ++r) {
      int o = (mt << 4) + half * 8 + r;
      float sc = bn.g[o] * rsqrtf(bn.v[o] + EPS);
      float y = (acc[r] - bn.m[o]) * sc + bn.b[o];
      Hout[o * 512 + col] = (_Float16)fmaxf(y, 0.0f);
    }
  }
}

// ---------------------------------------------------------------------------
// Fully fused EdgeConv block: gather(knn) + bn_in + 4x(conv+BN+ReLU) + max,
// one workgroup per (batch, 16-point tile). All activations live in LDS f16;
// nothing [B,2C,N,K]-shaped ever touches HBM.
// ---------------------------------------------------------------------------
template <int CIN2>
__global__ void __launch_bounds__(256) edge_block_kernel(EdgeArgs A) {
  extern __shared__ char smraw[];
  _Float16* F  = (_Float16*)smraw;               // [CIN2][512]
  _Float16* Ha = F + CIN2 * 512;                 // [64][512]
  _Float16* Hb = Ha + 64 * 512;                  // [64][512]
  int* sIdx = (int*)(Hb + 64 * 512);             // [512]

  int b = blockIdx.y, n0 = blockIdx.x * 16;
  int tid = threadIdx.x;
  for (int s = tid; s < 512; s += 256)
    sIdx[s] = A.idx[((long)b * 2048 + n0 + (s >> 5)) * 32 + (s & 31)];
  __syncthreads();

  const int C = CIN2 / 2;
  for (int e = tid; e < C * 512; e += 256) {
    int c = e >> 9, s = e & 511;
    int n = n0 + (s >> 5);
    const float* xc = A.xin + (long)b * A.xin_bstride + (long)c * 2048;
    float ctr = xc[n];
    float nbr = xc[sIdx[s]];
    float s1 = A.bnin.g[c] * rsqrtf(A.bnin.v[c] + EPS);
    F[c * 512 + s] = (_Float16)((nbr - ctr - A.bnin.m[c]) * s1 + A.bnin.b[c]);
    float s2 = A.bnin.g[C + c] * rsqrtf(A.bnin.v[C + c] + EPS);
    F[(C + c) * 512 + s] = (_Float16)((ctr - A.bnin.m[C + c]) * s2 + A.bnin.b[C + c]);
  }
  __syncthreads();

  conv_wmma(F,  Ha, A.w[0], CIN2, 1, 0, 32, 32, A.bn[0]);  __syncthreads();
  conv_wmma(Ha, Hb, A.w[1], 64,   2, 1, 32, 31, A.bn[1]);  __syncthreads();
  conv_wmma(Hb, Ha, A.w[2], 64,   4, 2, 31, 28, A.bn[2]);  __syncthreads();
  conv_wmma(Ha, Hb, A.w[3], 64,   8, 3, 28, 21, A.bn[3]);  __syncthreads();

  // max over neighbor axis (21 survivors)
  for (int e = tid; e < 64 * 16; e += 256) {
    int o = e >> 4, n = e & 15;
    float mx = -3.4e38f;
    for (int j = 0; j < 21; ++j) mx = fmaxf(mx, (float)Hb[o * 512 + n * 21 + j]);
    A.out[(long)b * A.out_bstride + (long)(A.out_coff + o) * 2048 + n0 + n] = mx;
    if (A.out16) A.out16[((long)b * 64 + o) * 2048 + n0 + n] = (_Float16)mx;
  }
}

// Dense layer (bias + BN + ReLU fused), 128 samples per workgroup, f16 LDS.
__device__ __forceinline__ void dense_wmma(const _Float16* __restrict__ Xin,
                                           _Float16* __restrict__ Xout,
                                           int CI, int CO,
                                           const float* __restrict__ W,
                                           const float* __restrict__ bias, BnP bn) {
  int tid = threadIdx.x, wave = tid >> 5, lane = tid & 31;
  int half = (lane >> 4) & 1, lm = lane & 15;
  int MT = CO >> 4, KB = CI >> 5;
  for (int t = wave; t < MT * 8; t += 8) {
    int mt = t >> 3, nt = t & 7;
    int col = (nt << 4) | lm;
    v8f acc = {};
    for (int kb = 0; kb < KB; ++kb) {
      v16h a, bf;
      int o = (mt << 4) | lm;
#pragma unroll
      for (int h = 0; h < 16; ++h) {
        int kk = (kb << 5) + half * 16 + h;
        a[h] = (_Float16)W[o * CI + kk];
      }
#pragma unroll
      for (int h = 0; h < 16; ++h) {
        int kk = (kb << 5) + half * 16 + h;
        bf[h] = Xin[kk * 128 + col];
      }
      acc = __builtin_amdgcn_wmma_f32_16x16x32_f16(false, a, false, bf,
                                                   (short)0, acc, false, false);
    }
#pragma unroll
    for (int r = 0; r < 8; ++r) {
      int o = (mt << 4) + half * 8 + r;
      float sc = bn.g[o] * rsqrtf(bn.v[o] + EPS);
      float y = (acc[r] + bias[o] - bn.m[o]) * sc + bn.b[o];
      Xout[o * 128 + col] = (_Float16)fmaxf(y, 0.0f);
    }
  }
}

// Classifier: 128->64->256->40, per-sample outputs to ws (no atomics).
__global__ void __launch_bounds__(256) classifier_kernel(ClsArgs A) {
  extern __shared__ char smraw[];
  _Float16* R0 = (_Float16*)smraw;               // [256][128]
  _Float16* R1 = R0 + 256 * 128;                 // [256][128]
  int tid = threadIdx.x;
  long s0 = (long)blockIdx.x * 128;
  int b = (int)(s0 >> 11), n0 = (int)(s0 & 2047);
  for (int e = tid; e < 128 * 128; e += 256) {
    int c = e >> 7, j = e & 127;
    R0[c * 128 + j] = (_Float16)A.cat[((long)b * 128 + c) * 2048 + n0 + j];
  }
  __syncthreads();
  dense_wmma(R0, R1, 128, 64,  A.w[0], A.bias[0], A.bn[0]);  __syncthreads();
  dense_wmma(R1, R0, 64,  256, A.w[1], A.bias[1], A.bn[1]);  __syncthreads();

  // last layer 256->40 (pad M to 48), write per-sample activations
  int wave = tid >> 5, lane = tid & 31, half = (lane >> 4) & 1, lm = lane & 15;
  for (int t = wave; t < 3 * 8; t += 8) {
    int mt = t >> 3, nt = t & 7;
    int col = (nt << 4) | lm;
    v8f acc = {};
    for (int kb = 0; kb < 8; ++kb) {
      v16h a, bf;
      int o = (mt << 4) | lm;
#pragma unroll
      for (int h = 0; h < 16; ++h) {
        int kk = (kb << 5) + half * 16 + h;
        a[h] = (o < 40) ? (_Float16)A.w[2][o * 256 + kk] : (_Float16)0.0f;
      }
#pragma unroll
      for (int h = 0; h < 16; ++h) {
        int kk = (kb << 5) + half * 16 + h;
        bf[h] = R0[kk * 128 + col];
      }
      acc = __builtin_amdgcn_wmma_f32_16x16x32_f16(false, a, false, bf,
                                                   (short)0, acc, false, false);
    }
#pragma unroll
    for (int r = 0; r < 8; ++r) {
      int o = (mt << 4) + half * 8 + r;
      if (o < 40) {
        float sc = A.bn[2].g[o] * rsqrtf(A.bn[2].v[o] + EPS);
        float y = (acc[r] + A.bias[2][o] - A.bn[2].m[o]) * sc + A.bn[2].b[o];
        A.y[(long)(b * 40 + o) * 2048 + n0 + col] = fmaxf(y, 0.0f);
      }
    }
  }
}

// Deterministic mean over N (fixed summation order).
__global__ void __launch_bounds__(256) mean_kernel(const float* __restrict__ y,
                                                   float* __restrict__ out) {
  int i = blockIdx.x * 256 + threadIdx.x;
  if (i >= 640) return;
  const float* p = y + (long)i * 2048;
  float s = 0.f;
  for (int j = 0; j < 2048; ++j) s += p[j];
  out[i] = s * (1.0f / 2048.0f);
}

// ---------------------------------------------------------------------------
extern "C" void kernel_launch(void* const* d_in, const int* in_sizes, int n_in,
                              void* d_out, int out_size, void* d_ws, size_t ws_size,
                              hipStream_t stream) {
  (void)in_sizes; (void)n_in; (void)out_size; (void)ws_size;
  // pytree order (sorted dict keys): c1.bn_in 0-3, c1.bns 4-19, c1.ws 20-23,
  // c2.bn_in 24-27, c2.bns 28-43, c2.ws 44-47, cls_bns 48-59, cls_bs 60-62,
  // cls_ws 63-65, x0 66.
  auto F = [&](int i) { return (const float*)d_in[i]; };
  auto BN = [&](int base) { return BnP{F(base), F(base + 1), F(base + 2), F(base + 3)}; };
  const float* x0 = F(66);

  // workspace layout
  char* w = (char*)d_ws;
  int*      idx1 = (int*)w;                              //  4 MB
  int*      idx2 = (int*)(w + (4ll << 20));              //  4 MB
  float*    cat  = (float*)(w + (8ll << 20));            // 16 MB  [16][128][2048]
  _Float16* x1h  = (_Float16*)(w + (24ll << 20));        //  4 MB  [16][64][2048]
  float*    xx   = (float*)(w + (28ll << 20));           // 128 KB [16][2048]
  float*    S    = (float*)(w + (29ll << 20));           // 16 MB  [2048][2048]
  float*    ycls = (float*)(w + (45ll << 20));           // ~5 MB  [16*40][2048]

  // 1) kNN on raw coords
  knn3_kernel<<<dim3(8, 16), 256, 0, stream>>>(x0, idx1);

  // 2) EdgeConv block 1 (2C=6) -> cat[:,0:64,:] and f16 copy for the Gram
  EdgeArgs e1;
  e1.xin = x0; e1.xin_bstride = 3 * 2048; e1.idx = idx1;
  e1.bnin = BN(0);
  for (int l = 0; l < 4; ++l) { e1.w[l] = F(20 + l); e1.bn[l] = BN(4 + 4 * l); }
  e1.out = cat; e1.out_bstride = 128 * 2048; e1.out_coff = 0; e1.out16 = x1h;
  edge_block_kernel<6><<<dim3(128, 16), 256, (6 + 128) * 512 * 2 + 2048, stream>>>(e1);

  // 3) kNN on x1: WMMA Gram per batch (16 MB, L2-resident) + top-32 select
  sumsq_kernel<<<dim3(128), 256, 0, stream>>>(cat, xx);
  for (int b = 0; b < 16; ++b) {
    gram_kernel<<<dim3(2048), 256, 0, stream>>>(x1h + (long)b * 64 * 2048, S);
    knn_from_gram<<<dim3(8), 256, 0, stream>>>(S, xx, b, idx2);
  }

  // 4) EdgeConv block 2 (2C=128) -> cat[:,64:128,:]
  EdgeArgs e2;
  e2.xin = cat; e2.xin_bstride = 128 * 2048; e2.idx = idx2;
  e2.bnin = BN(24);
  for (int l = 0; l < 4; ++l) { e2.w[l] = F(44 + l); e2.bn[l] = BN(28 + 4 * l); }
  e2.out = cat; e2.out_bstride = 128 * 2048; e2.out_coff = 64; e2.out16 = nullptr;
  edge_block_kernel<128><<<dim3(128, 16), 256, (128 + 128) * 512 * 2 + 2048, stream>>>(e2);

  // 5) classifier + deterministic mean
  ClsArgs cl;
  cl.cat = cat;
  for (int l = 0; l < 3; ++l) { cl.w[l] = F(63 + l); cl.bias[l] = F(60 + l); cl.bn[l] = BN(48 + 4 * l); }
  cl.y = ycls;
  classifier_kernel<<<dim3(256), 256, 2 * 256 * 128 * 2, stream>>>(cl);
  mean_kernel<<<dim3(3), 256, 0, stream>>>(ycls, (float*)d_out);
}
